// GPT_27444841021523
// MI455X (gfx1250) — compile-verified, matrix operand order checked by reading, not run on hardware
//
#include <hip/hip_runtime.h>
#include <hip/hip_bf16.h>
#include <math.h>

// GPT-2 config from the reference
#define TT 1024
#define CC 1024
#define HH 16
#define DD 64
#define LL 12
#define VV 32000

typedef __attribute__((ext_vector_type(16))) _Float16 v16h;
typedef __attribute__((ext_vector_type(8)))  _Float16 v8h;
typedef __attribute__((ext_vector_type(8)))  float    v8f;
typedef __attribute__((ext_vector_type(4)))  unsigned v4u;
typedef __attribute__((ext_vector_type(8)))  unsigned v8u;

__device__ __forceinline__ v16h hcat(v8h lo, v8h hi) {
  return __builtin_shufflevector(lo, hi, 0,1,2,3,4,5,6,7,8,9,10,11,12,13,14,15);
}
__device__ __forceinline__ v8f zero8() {
  v8f z;
#pragma unroll
  for (int i = 0; i < 8; ++i) z[i] = 0.0f;
  return z;
}
__device__ __forceinline__ float redmax16(float v) {
#pragma unroll
  for (int m = 1; m < 16; m <<= 1) v = fmaxf(v, __shfl_xor(v, m, 32));
  return v;
}
__device__ __forceinline__ float redsum16(float v) {
#pragma unroll
  for (int m = 1; m < 16; m <<= 1) v += __shfl_xor(v, m, 32);
  return v;
}
// Convert 8 consecutive fp32 in LDS to a v8h half-fragment.
__device__ __forceinline__ v8h cvt8(const float* p) {
  const float4 a = *(const float4*)p;
  const float4 b = *(const float4*)(p + 4);
  v8h r;
  r[0] = (_Float16)a.x; r[1] = (_Float16)a.y; r[2] = (_Float16)a.z; r[3] = (_Float16)a.w;
  r[4] = (_Float16)b.x; r[5] = (_Float16)b.y; r[6] = (_Float16)b.z; r[7] = (_Float16)b.w;
  return r;
}

// CDNA5 async global->LDS copies (ASYNCcnt-tracked, no VGPR data movement).
__device__ __forceinline__ void async_copy_b128(unsigned lds_dst, const float* src) {
  asm volatile("global_load_async_to_lds_b128 %0, %1, off" :: "v"(lds_dst), "v"(src) : "memory");
}
__device__ __forceinline__ void async_copy_b32(unsigned lds_dst, const float* src) {
  asm volatile("global_load_async_to_lds_b32 %0, %1, off" :: "v"(lds_dst), "v"(src) : "memory");
}
__device__ __forceinline__ void wait_async0() {
  asm volatile("s_wait_asynccnt 0x0" ::: "memory");
}

// ---------------------------------------------------------------------------
// Embedding: x[t][c] = wte[idx[t]][c] + wpe[t][c]
// ---------------------------------------------------------------------------
__global__ void embed_kernel(const int* __restrict__ idx, const float* __restrict__ wte,
                             const float* __restrict__ wpe, float* __restrict__ x) {
  const int t = blockIdx.x;
  const int c = threadIdx.x * 4;
  const int tok = idx[t];
  float4 a = *(const float4*)(wte + (size_t)tok * CC + c);
  const float4 p = *(const float4*)(wpe + (size_t)t * CC + c);
  a.x += p.x; a.y += p.y; a.z += p.z; a.w += p.w;
  *(float4*)(x + (size_t)t * CC + c) = a;
}

// ---------------------------------------------------------------------------
// LayerNorm over C=1024, one row per block (256 threads, 4 elems/thread)
// ---------------------------------------------------------------------------
__global__ void ln_kernel(const float* __restrict__ x, const float* __restrict__ w,
                          const float* __restrict__ b, float* __restrict__ out) {
  const int row = blockIdx.x;
  const int tid = threadIdx.x;
  const float4 v = *(const float4*)(x + (size_t)row * CC + tid * 4);
  __shared__ float rs[256];
  __shared__ float rq[256];
  rs[tid] = v.x + v.y + v.z + v.w;
  rq[tid] = v.x * v.x + v.y * v.y + v.z * v.z + v.w * v.w;
  __syncthreads();
  for (int s = 128; s > 0; s >>= 1) {
    if (tid < s) { rs[tid] += rs[tid + s]; rq[tid] += rq[tid + s]; }
    __syncthreads();
  }
  const float mean = rs[0] * (1.0f / CC);
  const float var  = rq[0] * (1.0f / CC) - mean * mean;
  const float rstd = rsqrtf(var + 1e-5f);
  const int c = tid * 4;
  const float4 wv = *(const float4*)(w + c);
  const float4 bv = *(const float4*)(b + c);
  float4 o;
  o.x = (v.x - mean) * rstd * wv.x + bv.x;
  o.y = (v.y - mean) * rstd * wv.y + bv.y;
  o.z = (v.z - mean) * rstd * wv.z + bv.z;
  o.w = (v.w - mean) * rstd * wv.w + bv.w;
  *(float4*)(out + (size_t)row * CC + c) = o;
}

// ---------------------------------------------------------------------------
// WMMA GEMM: Out[M,N] = epilogue(A[M,K] @ B[K,N] + bias[N])
//   op==0: bias only, op==1: bias + exact GELU, op==2: bias + residual add
// Block tile 128(M) x 64(N), BK=32; 8 waves, each wave owns a 16x64 strip.
//
// Double-buffered pipeline: the Tensor Data Mover streams the NEXT A tile
// (128x32 fp32, padded to a 36-DWORD LDS pitch via the D# pad feature) while
// the current tile is consumed by WMMAs; TENSORcnt in-order semantics let us
// wait "<=1" so one DMA stays in flight across the compute phase. B tiles are
// staged fp32->f16 transposed by the VALU (TDM cannot transpose), also double
// buffered, with global_prefetch two K-steps ahead.
// ---------------------------------------------------------------------------
#define AP 36   // LDS pitch of A tile in DWORDs (32 data + 4 pad)

__device__ __forceinline__ void tdm_issue_A(const float* A, int K, int m0, int kb,
                                            unsigned lds_addr, v8u g1) {
  const unsigned long long ga =
      (unsigned long long)(uintptr_t)A + (((size_t)m0 * K + kb) << 2);
  v4u g0;
  g0[0] = 1u;                                                 // count=1, user mode
  g0[1] = lds_addr;                                           // LDS byte address
  g0[2] = (unsigned)ga;                                       // global addr [31:0]
  g0[3] = ((unsigned)(ga >> 32) & 0x01FFFFFFu) | (2u << 30);  // addr[56:32], type=2
  asm volatile("tensor_load_to_lds %0, %1" :: "s"(g0), "s"(g1) : "memory");
}

__device__ __forceinline__ void stage_B(const float* __restrict__ B, int N, int n0, int kb,
                                        int tid, _Float16 (*sb)[40], bool pf) {
#pragma unroll
  for (int i = 0; i < 2; ++i) {
    const int f4 = tid + i * 256;        // 0..511
    const int kk = f4 >> 4;              // 0..31
    const int nq = (f4 & 15) << 2;       // 0,4,...,60
    const float* src = B + (size_t)(kb + kk) * N + n0 + nq;
    const float4 v = *(const float4*)src;
    if (pf) __builtin_prefetch(src + (size_t)32 * N, 0, 1);   // 2 K-steps ahead overall
    sb[nq + 0][kk] = (_Float16)v.x;
    sb[nq + 1][kk] = (_Float16)v.y;
    sb[nq + 2][kk] = (_Float16)v.z;
    sb[nq + 3][kk] = (_Float16)v.w;
  }
}

__global__ void gemm_wmma_kernel(const float* __restrict__ A, const float* __restrict__ B,
                                 const float* __restrict__ bias, const float* __restrict__ resid,
                                 float* __restrict__ Out, int M, int N, int K, int op) {
  __shared__ float    sAf[2][128 * AP]  __attribute__((aligned(16)));  // fp32 [m][k], pitch AP
  __shared__ _Float16 sB[2][64][40]     __attribute__((aligned(16)));  // f16  [n][k]

  const int tid  = threadIdx.x;
  const int wave = tid >> 5;
  const int lane = tid & 31;
  const int m0 = blockIdx.y * 128;
  const int n0 = blockIdx.x * 64;

  v8f acc[4];
#pragma unroll
  for (int j = 0; j < 4; ++j) acc[j] = zero8();

  const int lrow = lane & 15;
  const int khi  = (lane >> 4) << 3;   // 0 or 8

  // --- Tensor DMA descriptor group1 (invariant), per CDNA5 ISA ---
  // data_size=4B (code 2), pad_enable, pad_interval=32 DW (code 4),
  // pad_amount=4 DW (code 3); tensor = K x M, tile = 32 x 128.
  v8u g1;
  g1[0] = (2u << 16) | (1u << 20) | (4u << 22) | (3u << 25);
  g1[1] = ((unsigned)K & 0xFFFFu) << 16;
  g1[2] = ((unsigned)K >> 16) | (((unsigned)M & 0xFFFFu) << 16);
  g1[3] = ((unsigned)M >> 16) | (32u << 16);          // tile_dim0 = 32
  g1[4] = 128u;                                       // tile_dim1 = 128, tile_dim2 = 0
  g1[5] = (unsigned)K;                                // tensor_dim0_stride lo
  g1[6] = 0u;
  g1[7] = 0u;
  const unsigned ldsA0 = (unsigned)(uintptr_t)&sAf[0][0];
  const unsigned ldsA1 = (unsigned)(uintptr_t)&sAf[1][0];

  const int nk = K >> 5;

  // Prologue: start the pipeline on tile 0.
  if (tid < 32) tdm_issue_A(A, K, m0, 0, ldsA0, g1);
  stage_B(B, N, n0, 0, tid, sB[0], 32 < K);

  for (int it = 0; it < nk; ++it) {
    const int cur = it & 1;
    const int kb  = it << 5;
    __syncthreads();   // compute(it-1) finished -> "next" buffers are free

    const bool has_next = (it + 1) < nk;
    if (has_next) {
      const int nxt = cur ^ 1;
      if (tid < 32) tdm_issue_A(A, K, m0, kb + 32, nxt ? ldsA1 : ldsA0, g1);
      stage_B(B, N, n0, kb + 32, tid, sB[nxt], (kb + 64) < K);
    }
    if (tid < 32) {
      if (has_next) __builtin_amdgcn_s_wait_tensorcnt(1);  // tile(it) done, tile(it+1) in flight
      else          __builtin_amdgcn_s_wait_tensorcnt(0);
    }
    __syncthreads();   // TDM(cur) data + sB[cur] visible to all waves

    // A fragment: fp32 from LDS -> f16 in registers
    const int arow = (wave << 4) + lrow;
    const float* ap = &sAf[cur][arow * AP];
    const v16h af = hcat(cvt8(ap + khi), cvt8(ap + khi + 16));
#pragma unroll
    for (int j = 0; j < 4; ++j) {
      const int bcol = j * 16 + lrow;
      const v8h b_lo = *(const v8h*)&sB[cur][bcol][khi];
      const v8h b_hi = *(const v8h*)&sB[cur][bcol][khi + 16];
      acc[j] = __builtin_amdgcn_wmma_f32_16x16x32_f16(
          false, af, false, hcat(b_lo, b_hi), (short)0, acc[j], false, false);
    }
  }

  // Epilogue
  const int lgrp = (lane >> 4) << 3;       // 0 or 8 (row offset)
#pragma unroll
  for (int j = 0; j < 4; ++j) {
    const int n = n0 + j * 16 + lrow;
    const float bv = bias[n];
#pragma unroll
    for (int r = 0; r < 8; ++r) {
      const int m = m0 + (wave << 4) + r + lgrp;
      float val = acc[j][r] + bv;
      if (op == 1) {
        val = 0.5f * val * (1.0f + erff(val * 0.70710678118654752f));
      } else if (op == 2) {
        val += resid[(size_t)m * N + n];
      }
      Out[(size_t)m * N + n] = val;
    }
  }
}

// ---------------------------------------------------------------------------
// Flash attention (causal). qkv: [T][3C], per-head q|k|v slices of width D=64.
// Grid: (H, T/64); 128 threads = 4 waves; wave w handles query rows
// qb = qblk*64 + w*16. Streams keys in tiles of 32 with online softmax.
//
// K/V tiles are staged with GLOBAL_LOAD_ASYNC_TO_LDS (ASYNCcnt): K as per-lane
// B128 copies, V *transposed for free* via per-lane B32 LDS destinations.
// fp32->f16 conversion happens at fragment-build time (co-executes with WMMA).
// QK^T and PV both via v_wmma_f32_16x16x32_f16.
// ---------------------------------------------------------------------------
__global__ void attn_kernel(const float* __restrict__ qkv, float* __restrict__ y) {
  __shared__ float    sKf[32][68]    __attribute__((aligned(16)));  // fp32 [key n][d]
  __shared__ float    sVf[64][36]    __attribute__((aligned(16)));  // fp32 [d][key n]
  __shared__ _Float16 sP[4][16][40]  __attribute__((aligned(16)));  // per-wave P scratch

  const int h    = blockIdx.x;
  const int qblk = blockIdx.y;
  const int tid  = threadIdx.x;
  const int wave = tid >> 5;
  const int lane = tid & 31;
  const int lrow = lane & 15;
  const int khi  = (lane >> 4) << 3;     // 0 or 8
  const int rgrp = (lane >> 4) << 3;     // C-layout row offset
  const int qb   = qblk * 64 + wave * 16;
  const float qscale = 0.125f;           // 1/sqrt(D)

  const unsigned sKbase = (unsigned)(uintptr_t)&sKf[0][0];
  const unsigned sVbase = (unsigned)(uintptr_t)&sVf[0][0];

  // Load Q fragments (16x64 split into two 16x32 A fragments), scale folded in.
  v16h qf[2];
  {
    const float* qbase = qkv + (size_t)(qb + lrow) * (3 * CC) + h * DD;
#pragma unroll
    for (int s = 0; s < 2; ++s) {
      const float* p = qbase + s * 32;
      const float4 a0 = *(const float4*)(p + khi);
      const float4 a1 = *(const float4*)(p + khi + 4);
      const float4 b0 = *(const float4*)(p + khi + 16);
      const float4 b1 = *(const float4*)(p + khi + 20);
      v16h q;
      q[0]  = (_Float16)(a0.x * qscale); q[1]  = (_Float16)(a0.y * qscale);
      q[2]  = (_Float16)(a0.z * qscale); q[3]  = (_Float16)(a0.w * qscale);
      q[4]  = (_Float16)(a1.x * qscale); q[5]  = (_Float16)(a1.y * qscale);
      q[6]  = (_Float16)(a1.z * qscale); q[7]  = (_Float16)(a1.w * qscale);
      q[8]  = (_Float16)(b0.x * qscale); q[9]  = (_Float16)(b0.y * qscale);
      q[10] = (_Float16)(b0.z * qscale); q[11] = (_Float16)(b0.w * qscale);
      q[12] = (_Float16)(b1.x * qscale); q[13] = (_Float16)(b1.y * qscale);
      q[14] = (_Float16)(b1.z * qscale); q[15] = (_Float16)(b1.w * qscale);
      qf[s] = q;
    }
  }

  float m_i[8], l_i[8];
#pragma unroll
  for (int r = 0; r < 8; ++r) { m_i[r] = -INFINITY; l_i[r] = 0.0f; }
  v8f o[4];
#pragma unroll
  for (int j = 0; j < 4; ++j) o[j] = zero8();

  const int kend = qblk * 64 + 64;       // uniform for whole block (causal bound)
  for (int kb = 0; kb < kend; kb += 32) {
    __syncthreads();
    // Stage K tile (32 keys x 64 dims) via async B128 copies, layout [n][d].
#pragma unroll
    for (int i = 0; i < 4; ++i) {
      const int f4 = tid + i * 128;      // 0..511
      const int n  = f4 >> 4;            // 0..31
      const int dq = (f4 & 15) << 2;     // 0..60
      async_copy_b128(sKbase + (unsigned)(n * 68 + dq) * 4,
                      qkv + (size_t)(kb + n) * (3 * CC) + CC + h * DD + dq);
    }
    // Stage V tile transposed via async B32 (per-lane LDS dst does the transpose).
#pragma unroll
    for (int i = 0; i < 16; ++i) {
      const int e = tid + i * 128;       // 0..2047
      const int n = e >> 6;              // 0..31
      const int d = e & 63;              // 0..63 (contiguous reads per wavefront)
      async_copy_b32(sVbase + (unsigned)(d * 36 + n) * 4,
                     qkv + (size_t)(kb + n) * (3 * CC) + 2 * CC + h * DD + d);
    }
    wait_async0();
    __syncthreads();

    // S = Q @ K^T : two 16x16 score tiles (keys kb..kb+15 and kb+16..kb+31)
    v8f s0 = zero8(), s1 = zero8();
#pragma unroll
    for (int s = 0; s < 2; ++s) {
      const v16h bk0 = hcat(cvt8(&sKf[lrow][khi + s * 32]),
                            cvt8(&sKf[lrow][khi + 16 + s * 32]));
      s0 = __builtin_amdgcn_wmma_f32_16x16x32_f16(
          false, qf[s], false, bk0, (short)0, s0, false, false);
      const v16h bk1 = hcat(cvt8(&sKf[16 + lrow][khi + s * 32]),
                            cvt8(&sKf[16 + lrow][khi + 16 + s * 32]));
      s1 = __builtin_amdgcn_wmma_f32_16x16x32_f16(
          false, qf[s], false, bk1, (short)0, s1, false, false);
    }

    // Online softmax update (row stats shared across the 16-lane group)
    const int col0 = kb + lrow;
    const int col1 = col0 + 16;
#pragma unroll
    for (int r = 0; r < 8; ++r) {
      const int row = qb + r + rgrp;
      const float a0 = (col0 <= row) ? s0[r] : -1e30f;
      const float a1 = (col1 <= row) ? s1[r] : -1e30f;
      const float tmax = redmax16(fmaxf(a0, a1));
      const float mnew = fmaxf(m_i[r], tmax);
      const float p0 = __expf(a0 - mnew);
      const float p1 = __expf(a1 - mnew);
      const float alpha = __expf(m_i[r] - mnew);
      const float psum = redsum16(p0 + p1);
      l_i[r] = l_i[r] * alpha + psum;
      m_i[r] = mnew;
      sP[wave][r + rgrp][lrow]      = (_Float16)p0;
      sP[wave][r + rgrp][16 + lrow] = (_Float16)p1;
#pragma unroll
      for (int j = 0; j < 4; ++j) o[j][r] *= alpha;
    }

    // O += P @ V
    const v8h pl = *(const v8h*)&sP[wave][lrow][khi];
    const v8h ph = *(const v8h*)&sP[wave][lrow][khi + 16];
    const v16h pf = hcat(pl, ph);
#pragma unroll
    for (int j = 0; j < 4; ++j) {
      const int d = j * 16 + lrow;
      const v16h vf = hcat(cvt8(&sVf[d][khi]), cvt8(&sVf[d][khi + 16]));
      o[j] = __builtin_amdgcn_wmma_f32_16x16x32_f16(
          false, pf, false, vf, (short)0, o[j], false, false);
    }
  }

  // Normalize and write y[T][C] at this head's slice
#pragma unroll
  for (int j = 0; j < 4; ++j) {
    const int d = j * 16 + lrow;
#pragma unroll
    for (int r = 0; r < 8; ++r) {
      const int row = qb + r + rgrp;
      y[(size_t)row * CC + h * DD + d] = o[j][r] / l_i[r];
    }
  }
}

// ---------------------------------------------------------------------------
// lm_head GEMV: out[v] = dot(xlast, wte[v]).  One wave per vocab row.
// ---------------------------------------------------------------------------
__global__ void logits_kernel(const float* __restrict__ xl, const float* __restrict__ wte,
                              float* __restrict__ out) {
  __shared__ float sx[CC];
  const int tid = threadIdx.x;
  ((float4*)sx)[tid] = ((const float4*)xl)[tid];   // 256 * 16B = 4KB
  __syncthreads();
  const int wave = tid >> 5;
  const int lane = tid & 31;
  const int row  = blockIdx.x * 8 + wave;
  const float* wr = wte + (size_t)row * CC;
  float acc = 0.0f;
#pragma unroll
  for (int k = lane * 4; k < CC; k += 128) {
    const float4 wv = *(const float4*)(wr + k);
    acc += wv.x * sx[k] + wv.y * sx[k + 1] + wv.z * sx[k + 2] + wv.w * sx[k + 3];
  }
#pragma unroll
  for (int m = 1; m < 32; m <<= 1) acc += __shfl_xor(acc, m, 32);
  if (lane == 0) out[row] = acc;
}

// ---------------------------------------------------------------------------
// Host-side orchestration
// ---------------------------------------------------------------------------
extern "C" void kernel_launch(void* const* d_in, const int* in_sizes, int n_in,
                              void* d_out, int out_size, void* d_ws, size_t ws_size,
                              hipStream_t stream) {
  const int*   idx    = (const int*)d_in[0];
  const float* wte    = (const float*)d_in[1];
  const float* wpe    = (const float*)d_in[2];
  const float* ln1_w  = (const float*)d_in[3];
  const float* ln1_b  = (const float*)d_in[4];
  const float* attn_w = (const float*)d_in[5];
  const float* attn_b = (const float*)d_in[6];
  const float* proj_w = (const float*)d_in[7];
  const float* proj_b = (const float*)d_in[8];
  const float* ln2_w  = (const float*)d_in[9];
  const float* ln2_b  = (const float*)d_in[10];
  const float* fc_w   = (const float*)d_in[11];
  const float* fc_b   = (const float*)d_in[12];
  const float* mlp_w  = (const float*)d_in[13];
  const float* mlp_b  = (const float*)d_in[14];
  const float* lnf_w  = (const float*)d_in[15];
  const float* lnf_b  = (const float*)d_in[16];

  float* x    = (float*)d_ws;                 // [T][C]
  float* hbuf = x    + (size_t)TT * CC;       // [T][C]
  float* qkv  = hbuf + (size_t)TT * CC;       // [T][3C]
  float* ybuf = qkv  + (size_t)TT * 3 * CC;   // [T][C]
  float* mbuf = ybuf + (size_t)TT * CC;       // [T][4C]
  float* xl   = mbuf + (size_t)TT * 4 * CC;   // [C]

  embed_kernel<<<TT, 256, 0, stream>>>(idx, wte, wpe, x);

  for (int l = 0; l < LL; ++l) {
    ln_kernel<<<TT, 256, 0, stream>>>(x, ln1_w + (size_t)l * CC, ln1_b + (size_t)l * CC, hbuf);
    gemm_wmma_kernel<<<dim3(3 * CC / 64, TT / 128), 256, 0, stream>>>(
        hbuf, attn_w + (size_t)l * CC * 3 * CC, attn_b + (size_t)l * 3 * CC,
        nullptr, qkv, TT, 3 * CC, CC, 0);
    attn_kernel<<<dim3(HH, TT / 64), 128, 0, stream>>>(qkv, ybuf);
    gemm_wmma_kernel<<<dim3(CC / 64, TT / 128), 256, 0, stream>>>(
        ybuf, proj_w + (size_t)l * CC * CC, proj_b + (size_t)l * CC,
        x, x, TT, CC, CC, 2);
    ln_kernel<<<TT, 256, 0, stream>>>(x, ln2_w + (size_t)l * CC, ln2_b + (size_t)l * CC, hbuf);
    gemm_wmma_kernel<<<dim3(4 * CC / 64, TT / 128), 256, 0, stream>>>(
        hbuf, fc_w + (size_t)l * CC * 4 * CC, fc_b + (size_t)l * 4 * CC,
        nullptr, mbuf, TT, 4 * CC, CC, 1);
    gemm_wmma_kernel<<<dim3(CC / 64, TT / 128), 256, 0, stream>>>(
        mbuf, mlp_w + (size_t)l * 4 * CC * CC, mlp_b + (size_t)l * CC,
        x, x, TT, CC, 4 * CC, 2);
  }

  // Final LN on the last row only (logits are last-position only)
  ln_kernel<<<1, 256, 0, stream>>>(x + (size_t)(TT - 1) * CC, lnf_w, lnf_b, xl);
  logits_kernel<<<VV / 8, 256, 0, stream>>>(xl, wte, (float*)d_out);
}
